// MedicalHGT_10170482557469
// MI455X (gfx1250) — matrix operand order repaired and checked by reading, not compiled, not run on hardware
//
#include <hip/hip_runtime.h>
#include <cmath>

#define H 2
#define D 32
#define HID 64
#define DIN 128
#define NQ 100000
#define NA 200000
#define NK 200000
#define EP_N 100000
#define EN_N 100000
#define NMAX 200000
#define EMAX 800000

typedef __attribute__((ext_vector_type(16))) __bf16 v16bf;
typedef __attribute__((ext_vector_type(8)))  float  v8f;

__device__ __forceinline__ float gelu_exact(float x) {
    return 0.5f * x * (1.0f + erff(x * 0.7071067811865475f));
}

__global__ void fill_kernel(float* p, float v, int n) {
    int i = blockIdx.x * blockDim.x + threadIdx.x;
    if (i < n) p[i] = v;
}

// Fuse per-relation head transforms into the projection weights:
//   Wf[r][i][h*32+e] = sum_d w[st(r)][i][h*32+d] * rel[r][h][d][e]
//   bf[r][h*32+e]    = sum_d b[st(r)][h*32+d]    * rel[r][h][d][e]
__global__ void fuse_w_kernel(const float* __restrict__ w,   // [3,64,64]
                              const float* __restrict__ b,   // [3,64]
                              const float* __restrict__ rel, // [4,2,32,32]
                              float* __restrict__ Wf,        // [4,64,64]
                              float* __restrict__ bf)        // [4,64]
{
    int idx = blockIdx.x * blockDim.x + threadIdx.x;
    if (idx >= 4 * 65 * 64) return;
    int r   = idx / (65 * 64);
    int rem = idx - r * (65 * 64);
    int i   = rem >> 6;        // 0..64 (64 == bias row)
    int c   = rem & 63;        // h*32+e
    int h   = c >> 5, e = c & 31;
    int st  = (r == 0) ? 2 : ((r == 3) ? 1 : 0);
    const float* relp = rel + (size_t)((r * H + h) * D) * D + e; // stride D over d
    float acc = 0.f;
    if (i < 64) {
        const float* wp = w + (size_t)(st * HID + i) * HID + h * 32;
        for (int d = 0; d < 32; ++d) acc += wp[d] * relp[(size_t)d * D];
        Wf[(size_t)(r * HID + i) * HID + c] = acc;
    } else {
        const float* bp = b + st * HID + h * 32;
        for (int d = 0; d < 32; ++d) acc += bp[d] * relp[(size_t)d * D];
        bf[r * HID + c] = acc;
    }
}

// Skinny GEMM, N fixed at 64: C[M,64] = post( act_in(A[M,K]) @ W[K,64] + bias )
// Weights staged once per block into LDS, transposed to [64][K] bf16 so each
// B fragment is two contiguous ds_load_b128. One wave computes a 16-row tile;
// all K fragments of A hoisted into registers; bf16 WMMA, f32 accumulate.
template<int K, bool GELU_IN, bool RELU_OUT, bool BLEND>
__global__ __launch_bounds__(128)
void gemm_kernel(const float* __restrict__ A, int lda,
                 const float* __restrict__ W, int ldw,
                 const float* __restrict__ bias,
                 float* __restrict__ C, int ldc,
                 int M,
                 const float* __restrict__ resid,  // skip blend (BLEND only)
                 const float* __restrict__ skipp)  // beta = sigmoid(*skipp)
{
    constexpr int KS = K / 32;
    constexpr int N  = 64;
    __shared__ __align__(32) __bf16 wlds[N * K];   // W transposed: [n][k]

    // cooperative stage+transpose (coalesced global reads over n)
    for (int t = threadIdx.x; t < N * K; t += 128) {
        int k = t >> 6, n = t & 63;
        wlds[n * K + k] = (__bf16)W[(size_t)k * ldw + n];
    }
    __syncthreads();

    const int wid  = threadIdx.x >> 5;
    const int lane = threadIdx.x & 31;
    const int m0   = blockIdx.x * 64 + wid * 16;
    if (m0 >= M) return;                // whole-wave exit: EXEC stays all-ones
    const int g = lane >> 4;            // lane group
    const int r = lane & 15;

    // A fragments for every K step (16-bit A 16x32, contiguous k: group g holds k 16g..16g+15)
    v16bf afr[KS];
    const float* Arow = A + (size_t)(m0 + r) * lda;
    __builtin_prefetch(Arow + (size_t)64 * lda, 0, 1);   // next block's tile
#pragma unroll
    for (int ks = 0; ks < KS; ++ks) {
        const int kb = ks * 32 + g * 16;
        float4 r0 = *(const float4*)(Arow + kb);
        float4 r1 = *(const float4*)(Arow + kb + 4);
        float4 r2 = *(const float4*)(Arow + kb + 8);
        float4 r3 = *(const float4*)(Arow + kb + 12);
        float t[16] = {r0.x, r0.y, r0.z, r0.w, r1.x, r1.y, r1.z, r1.w,
                       r2.x, r2.y, r2.z, r2.w, r3.x, r3.y, r3.z, r3.w};
        v16bf af;
#pragma unroll
        for (int j = 0; j < 16; ++j) {
            float x = t[j];
            if (GELU_IN) x = gelu_exact(x);
            af[j] = (__bf16)x;
        }
        afr[ks] = af;
    }

    float beta = 1.0f, ombeta = 0.0f;
    if (BLEND) {
        float sk = *skipp;
        beta   = 1.0f / (1.0f + __expf(-sk));
        ombeta = 1.0f - beta;
    }

#pragma unroll
    for (int n0 = 0; n0 < N; n0 += 16) {
        const int col = n0 + r;
        const __bf16* wcol = &wlds[col * K];
        v8f acc = {};
#pragma unroll
        for (int ks = 0; ks < KS; ++ks) {
            // 16-bit B 32x16: lane = column, 16 consecutive K, lane-group = K half
            const int kb = ks * 32 + g * 16;
            v16bf bf = *(const v16bf*)(wcol + kb);      // 2x ds_load_b128
            acc = __builtin_amdgcn_wmma_f32_16x16x32_bf16(
                false, afr[ks], false, bf, (short)0, acc, false, false);
        }
        const float bcol = bias[col];
#pragma unroll
        for (int j = 0; j < 8; ++j) {
            const int rowo = m0 + j + (g ? 8 : 0);
            float v = acc[j] + bcol;
            if (RELU_OUT) v = fmaxf(v, 0.0f);
            if (BLEND) v = beta * v + ombeta * resid[(size_t)rowo * ldc + col];
            C[(size_t)rowo * ldc + col] = v;
        }
    }
}

__device__ __forceinline__ void atomicMaxF(float* addr, float val) {
    if (val >= 0.f) atomicMax((int*)addr, __float_as_int(val));
    else            atomicMin((unsigned int*)addr, (unsigned int)__float_as_int(val));
}

// alpha[e,h] = (q[dst] . k_rel[src]) * p_rel[h] * 1/sqrt(D); seg-max into mbuf
__global__ void edge_alpha_kernel(const float* __restrict__ qbuf,
                                  const float* __restrict__ krel,
                                  const int* __restrict__ es, const int* __restrict__ ed,
                                  const float* __restrict__ prel,
                                  float* __restrict__ alpha, float* __restrict__ mbuf,
                                  int E)
{
    int idx = blockIdx.x * blockDim.x + threadIdx.x;
    if (idx >= E * H) return;
    int e = idx >> 1, h = idx & 1;
    int s = es[e], d = ed[e];
    const float4* qp = (const float4*)(qbuf + (size_t)d * HID + h * D);
    const float4* kp = (const float4*)(krel + (size_t)s * HID + h * D);
    float acc = 0.f;
#pragma unroll
    for (int j = 0; j < 8; ++j) {
        float4 a = qp[j], b = kp[j];
        acc += a.x * b.x + a.y * b.y + a.z * b.z + a.w * b.w;
    }
    float al = acc * prel[h] * 0.17677669529663687f;  // 1/sqrt(32)
    alpha[idx] = al;
    atomicMaxF(mbuf + d * H + h, al);
}

__global__ void edge_expsum_kernel(const float* __restrict__ alpha,
                                   const int* __restrict__ ed,
                                   const float* __restrict__ mbuf,
                                   float* __restrict__ expE, float* __restrict__ sbuf,
                                   int E)
{
    int idx = blockIdx.x * blockDim.x + threadIdx.x;
    if (idx >= E * H) return;
    int e = idx >> 1, h = idx & 1;
    int d = ed[e];
    float ev = __expf(alpha[idx] - mbuf[d * H + h]);
    expE[idx] = ev;
    atomicAdd(sbuf + d * H + h, ev);
}

__global__ void edge_scatter_kernel(const float* __restrict__ expE,
                                    const float* __restrict__ sbuf,
                                    const float* __restrict__ vrel,
                                    const int* __restrict__ es, const int* __restrict__ ed,
                                    float* __restrict__ outp, int E)
{
    int idx = blockIdx.x * blockDim.x + threadIdx.x;
    if (idx >= E * H) return;
    int e = idx >> 1, h = idx & 1;
    int s = es[e], d = ed[e];
    float coeff = expE[idx] / (sbuf[d * H + h] + 1e-16f);
    const float4* vp = (const float4*)(vrel + (size_t)s * HID + h * D);
    float* op = outp + (size_t)d * HID + h * D;
#pragma unroll
    for (int j = 0; j < 8; ++j) {
        float4 v = vp[j];
        atomicAdd(op + 4 * j + 0, coeff * v.x);
        atomicAdd(op + 4 * j + 1, coeff * v.y);
        atomicAdd(op + 4 * j + 2, coeff * v.z);
        atomicAdd(op + 4 * j + 3, coeff * v.w);
    }
}

__global__ void pred_kernel(const float* __restrict__ zq, const float* __restrict__ za,
                            const int* __restrict__ src, const int* __restrict__ dst,
                            float* __restrict__ outp, int E)
{
    int e = blockIdx.x * blockDim.x + threadIdx.x;
    if (e >= E) return;
    const float4* a = (const float4*)(zq + (size_t)src[e] * HID);
    const float4* b = (const float4*)(za + (size_t)dst[e] * HID);
    float acc = 0.f;
#pragma unroll
    for (int j = 0; j < 16; ++j) {
        float4 x = a[j], y = b[j];
        acc += x.x * y.x + x.y * y.y + x.z * y.z + x.w * y.w;
    }
    outp[e] = 1.0f / (1.0f + __expf(-acc));
}

extern "C" void kernel_launch(void* const* d_in, const int* in_sizes, int n_in,
                              void* d_out, int out_size, void* d_ws, size_t ws_size,
                              hipStream_t stream)
{
    (void)in_sizes; (void)n_in; (void)out_size; (void)ws_size;

    const float* xin[3] = {(const float*)d_in[0], (const float*)d_in[1], (const float*)d_in[2]};
    const float* lin_w = (const float*)d_in[3];
    const float* lin_b = (const float*)d_in[4];
    const float* k_w   = (const float*)d_in[5];
    const float* k_b   = (const float*)d_in[6];
    const float* q_w   = (const float*)d_in[7];
    const float* q_b   = (const float*)d_in[8];
    const float* v_w   = (const float*)d_in[9];
    const float* v_b   = (const float*)d_in[10];
    const float* a_w   = (const float*)d_in[11];
    const float* a_b   = (const float*)d_in[12];
    const float* a_rel = (const float*)d_in[13];
    const float* m_rel = (const float*)d_in[14];
    const float* p_rel = (const float*)d_in[15];
    const float* skip  = (const float*)d_in[16];

    const int ns[3]    = {NQ, NA, NK};
    const int st_of[4] = {2, 0, 0, 1};
    const int dt_of[4] = {0, 2, 1, 0};
    const int Es[4]    = {800000, 800000, 400000, 400000};
    const int* srcs[4] = {(const int*)d_in[17], (const int*)d_in[19],
                          (const int*)d_in[21], (const int*)d_in[23]};
    const int* dsts[4] = {(const int*)d_in[18], (const int*)d_in[20],
                          (const int*)d_in[22], (const int*)d_in[24]};

    // ---- workspace layout (floats) ----
    float* ws = (float*)d_ws;
    size_t off = 0;
    float* xs[3];   for (int i = 0; i < 3; ++i) { xs[i]   = ws + off; off += (size_t)ns[i] * HID; }
    float* outb[3]; for (int i = 0; i < 3; ++i) { outb[i] = ws + off; off += (size_t)ns[i] * HID; }
    float* qbuf = ws + off; off += (size_t)NMAX * HID;
    float* krel = ws + off; off += (size_t)NMAX * HID;
    float* vrel = ws + off; off += (size_t)NMAX * HID;
    float* Wfk  = ws + off; off += 4 * HID * HID;
    float* Wfv  = ws + off; off += 4 * HID * HID;
    float* bfk  = ws + off; off += 4 * HID;
    float* bfv  = ws + off; off += 4 * HID;
    float* alphaB = ws + off; off += (size_t)EMAX * H;
    float* expB   = ws + off; off += (size_t)EMAX * H;
    float* mbuf   = ws + off; off += (size_t)NMAX * H;
    float* sbuf   = ws + off; off += (size_t)NMAX * H;

    // ---- fused relation weights ----
    {
        int tot = 4 * 65 * 64;
        fuse_w_kernel<<<(tot + 255) / 256, 256, 0, stream>>>(k_w, k_b, a_rel, Wfk, bfk);
        fuse_w_kernel<<<(tot + 255) / 256, 256, 0, stream>>>(v_w, v_b, m_rel, Wfv, bfv);
    }

    // ---- xs[i] = relu(x @ lin_w[i] + lin_b[i]) ----
    for (int i = 0; i < 3; ++i) {
        gemm_kernel<DIN, false, true, false><<<dim3((ns[i] + 63) / 64), 128, 0, stream>>>(
            xin[i], DIN, lin_w + (size_t)i * DIN * HID, HID, lin_b + i * HID,
            xs[i], HID, ns[i], nullptr, nullptr);
    }

    // ---- zero out-accumulators (contiguous) ----
    {
        int tot = (NQ + NA + NK) * HID;
        fill_kernel<<<(tot + 255) / 256, 256, 0, stream>>>(outb[0], 0.f, tot);
    }

    // ---- per-relation attention ----
    for (int r = 0; r < 4; ++r) {
        int st = st_of[r], dt = dt_of[r], E = Es[r];
        gemm_kernel<HID, false, false, false><<<dim3((ns[dt] + 63) / 64), 128, 0, stream>>>(
            xs[dt], HID, q_w + (size_t)dt * HID * HID, HID, q_b + dt * HID,
            qbuf, HID, ns[dt], nullptr, nullptr);
        gemm_kernel<HID, false, false, false><<<dim3((ns[st] + 63) / 64), 128, 0, stream>>>(
            xs[st], HID, Wfk + (size_t)r * HID * HID, HID, bfk + r * HID,
            krel, HID, ns[st], nullptr, nullptr);
        gemm_kernel<HID, false, false, false><<<dim3((ns[st] + 63) / 64), 128, 0, stream>>>(
            xs[st], HID, Wfv + (size_t)r * HID * HID, HID, bfv + r * HID,
            vrel, HID, ns[st], nullptr, nullptr);

        int nm = ns[dt] * H;
        fill_kernel<<<(nm + 255) / 256, 256, 0, stream>>>(mbuf, -INFINITY, nm);
        fill_kernel<<<(nm + 255) / 256, 256, 0, stream>>>(sbuf, 0.f, nm);

        int tot = E * H;
        edge_alpha_kernel<<<(tot + 255) / 256, 256, 0, stream>>>(
            qbuf, krel, srcs[r], dsts[r], p_rel + r * H, alphaB, mbuf, E);
        edge_expsum_kernel<<<(tot + 255) / 256, 256, 0, stream>>>(
            alphaB, dsts[r], mbuf, expB, sbuf, E);
        edge_scatter_kernel<<<(tot + 255) / 256, 256, 0, stream>>>(
            expB, sbuf, vrel, srcs[r], dsts[r], outb[dt], E);
    }

    // ---- z[i] = sigmoid(skip)*(gelu(out) @ a_w + a_b) + (1-sigmoid)*xs ----
    float* zq = (float*)d_out + EP_N + EN_N;
    float* za = zq + (size_t)NQ * HID;
    float* zk = za + (size_t)NA * HID;
    float* z[3] = {zq, za, zk};
    for (int i = 0; i < 3; ++i) {
        gemm_kernel<HID, true, false, true><<<dim3((ns[i] + 63) / 64), 128, 0, stream>>>(
            outb[i], HID, a_w + (size_t)i * HID * HID, HID, a_b + i * HID,
            z[i], HID, ns[i], xs[i], skip + i);
    }

    // ---- link predictions ----
    pred_kernel<<<(EP_N + 255) / 256, 256, 0, stream>>>(
        zq, za, (const int*)d_in[25], (const int*)d_in[26], (float*)d_out, EP_N);
    pred_kernel<<<(EN_N + 255) / 256, 256, 0, stream>>>(
        zq, za, (const int*)d_in[27], (const int*)d_in[28], (float*)d_out + EP_N, EN_N);
}
